// Model_43293270344322
// MI455X (gfx1250) — matrix operand order repaired and checked by reading, not compiled, not run on hardware
//
#include <hip/hip_runtime.h>
#include <cstdint>

#define VOCAB 50000
#define D     1024
#define BATCH 512
#define SEQL  512
#define BT    16            // batch rows per workgroup (one WMMA M-tile)
#define HPAD  1032          // h row stride in halfs: 2064B -> +4 dword skew, no LDS bank conflicts

typedef __attribute__((ext_vector_type(16))) _Float16 v16h;
typedef __attribute__((ext_vector_type(8)))  _Float16 v8h;
typedef __attribute__((ext_vector_type(4)))  _Float16 v4h;
typedef __attribute__((ext_vector_type(8)))  float    v8f;

#define AS1 __attribute__((address_space(1)))
#define AS3 __attribute__((address_space(3)))

// fast tanh: 1 - 2/(e^{2x}+1); saturates correctly at +/-1, no NaN paths
__device__ __forceinline__ float fast_tanh(float x) {
  float e = __expf(2.0f * x);
  return 1.0f - 2.0f * __builtin_amdgcn_rcpf(e + 1.0f);
}

// ---------------------------------------------------------------------------
// Kernel 1: one-time f32 -> f16 conversion of V_w into workspace (2 MB).
// Layout preserved: Vh[n][k] = (f16)V_w[n][k], row-major, row stride D.
// ---------------------------------------------------------------------------
__global__ __launch_bounds__(256) void convert_vw(const float* __restrict__ Vw,
                                                  _Float16* __restrict__ Vh) {
  int i = (blockIdx.x * 256 + threadIdx.x) * 4;
  float4 f = *(const float4*)(Vw + i);
  v4h h;
  h[0] = (_Float16)f.x; h[1] = (_Float16)f.y;
  h[2] = (_Float16)f.z; h[3] = (_Float16)f.w;
  *(v4h*)(Vh + i) = h;
}

// ---------------------------------------------------------------------------
// Kernel 2: fused 512-step Elman scan + head + softmax.
// grid = BATCH/BT = 32 blocks, block = 512 threads = 16 waves (wave32).
// Each wave owns 4 N-tiles (64 columns). h kept in LDS as f16, double-barrier
// per step. WMMA: D(f32 16x16) += A(h 16x32 f16) * B(V_w^T 32x16 f16).
// ---------------------------------------------------------------------------
__global__ __launch_bounds__(512) void rnn_scan(
    const int*   __restrict__ xp,     // [BATCH][SEQL] token ids
    const float* __restrict__ Uemb,   // [VOCAB][D]
    const float* __restrict__ Vb,     // [D]
    const _Float16* __restrict__ Vh,  // [D][D] f16 V_w
    const float* __restrict__ Ww,     // [2][D]
    const float* __restrict__ Wb,     // [2]
    float*       __restrict__ out) {  // [BATCH][2] softmax

  __shared__ _Float16 h_sh[BT][HPAD];
  __shared__ int      tok_sh[2][BT];

  const int tid  = threadIdx.x;
  const int wave = tid >> 5;
  const int lane = tid & 31;
  const int lhi  = lane >> 4;   // 0 / 1 (half-wave)
  const int llo  = lane & 15;
  const int b0   = blockIdx.x * BT;

  // zero hidden state
  for (int i = tid; i < BT * HPAD; i += 512) ((_Float16*)h_sh)[i] = (_Float16)0.f;
  if (tid < BT) tok_sh[0][tid] = xp[(size_t)(b0 + tid) * SEQL];

  // per-wave constants: 4 output N-tiles + their bias values
  int   n0[4];
  float vb[4];
#pragma unroll
  for (int j = 0; j < 4; ++j) {
    n0[j] = wave * 64 + j * 16;
    vb[j] = Vb[n0[j] + llo];
  }
  const int m_a  = llo;       // A-fragment row (M)
  const int koff = lhi * 16;  // A-fragment K sub-offset (contiguous k = 16h + s, matches B)
  const int mrow = lhi * 8;   // C/D fragment M base (M = mrow + r)

  for (int t = 0; t < SEQL; ++t) {
    __syncthreads();  // h writes from t-1 and tok[t] now visible

    const int* tok = tok_sh[t & 1];
    int tkr[8];
#pragma unroll
    for (int r = 0; r < 8; ++r) tkr[r] = tok[mrow + r];

    // C init: emb gather + bias (f32, straight into the accumulator)
    v8f acc[4];
#pragma unroll
    for (int j = 0; j < 4; ++j) {
#pragma unroll
      for (int r = 0; r < 8; ++r) {
        acc[j][r] = Uemb[(size_t)tkr[r] * D + n0[j] + llo] + vb[j];
      }
    }

    // K reduction: 128 WMMAs per wave per step (4 N-tiles x 32 K-steps)
#pragma unroll 2
    for (int k0 = 0; k0 < D; k0 += 32) {
      union { v16h v; v8h h[2]; } a;
      const _Float16* ap = &h_sh[m_a][k0 + koff];
      a.h[0] = *(const v8h*)ap;           // K = k0+koff .. +7     (ds_load_b128)
      a.h[1] = *(const v8h*)(ap + 8);     // K = k0+koff+8 .. +15  (ds_load_b128)

      // issue all 8 B global_load_b128s before the first WMMA (deep load-ahead)
      v16h bfrag[4];
#pragma unroll
      for (int j = 0; j < 4; ++j) {
        bfrag[j] = *(const v16h*)(Vh + (size_t)(n0[j] + llo) * D + k0 + lhi * 16);
      }

      // prefetch next 128B cacheline of each B row stream (WGP scope)
      if ((k0 & 32) == 0 && k0 + 64 < D) {
#pragma unroll
        for (int j = 0; j < 4; ++j)
          __builtin_prefetch(Vh + (size_t)(n0[j] + llo) * D + k0 + 64 + lhi * 16, 0, 3);
      }

#pragma unroll
      for (int j = 0; j < 4; ++j) {
        acc[j] = __builtin_amdgcn_wmma_f32_16x16x32_f16(
            /*neg_a=*/false, a.v, /*neg_b=*/false, bfrag[j],
            /*c_mod=*/(short)0, acc[j], /*reuse_a=*/false, /*reuse_b=*/false);
      }
    }

    // async prefetch of next step's token ids into LDS (CDNA5 async-to-LDS path)
    {
      const int tn = (t + 1 < SEQL) ? (t + 1) : t;
#if __has_builtin(__builtin_amdgcn_global_load_async_to_lds_b32)
      if (tid < BT) {
        int* gsrc = const_cast<int*>(xp + (size_t)(b0 + tid) * SEQL + tn);
        int* ldst = &tok_sh[(t + 1) & 1][tid];
        __builtin_amdgcn_global_load_async_to_lds_b32(
            (AS1 int*)gsrc, (AS3 int*)ldst, 0, 0);
      }
#if __has_builtin(__builtin_amdgcn_s_wait_asynccnt)
      __builtin_amdgcn_s_wait_asynccnt(0);
#endif
#else
      if (tid < BT)
        tok_sh[(t + 1) & 1][tid] = xp[(size_t)(b0 + tid) * SEQL + tn];
#endif
    }

    __syncthreads();  // all waves done reading h for step t

    // h' = tanh(z), write back as f16 (each (row,col) owned by exactly one lane)
#pragma unroll
    for (int j = 0; j < 4; ++j) {
#pragma unroll
      for (int r = 0; r < 8; ++r) {
        h_sh[mrow + r][n0[j] + llo] = (_Float16)fast_tanh(acc[j][r]);
      }
    }
  }

  __syncthreads();

  // head: one wave per batch row; lane-strided partial dot + wave32 reduction
  {
    const int row = wave;   // 16 waves == BT rows
    float p0 = 0.0f, p1 = 0.0f;
    for (int k = lane; k < D; k += 32) {
      float hv = (float)h_sh[row][k];
      p0 += hv * Ww[k];
      p1 += hv * Ww[D + k];
    }
#pragma unroll
    for (int off = 16; off > 0; off >>= 1) {
      p0 += __shfl_down(p0, off, 32);
      p1 += __shfl_down(p1, off, 32);
    }
    if (lane == 0) {
      float l0 = p0 + Wb[0];
      float l1 = p1 + Wb[1];
      float mx = fmaxf(l0, l1);
      float e0 = __expf(l0 - mx), e1 = __expf(l1 - mx);
      float inv = 1.0f / (e0 + e1);
      out[(size_t)(b0 + row) * 2 + 0] = e0 * inv;
      out[(size_t)(b0 + row) * 2 + 1] = e1 * inv;
    }
  }
}

// ---------------------------------------------------------------------------
extern "C" void kernel_launch(void* const* d_in, const int* in_sizes, int n_in,
                              void* d_out, int out_size, void* d_ws, size_t ws_size,
                              hipStream_t stream) {
  const int*   x    = (const int*)d_in[0];
  const float* Uemb = (const float*)d_in[1];
  const float* Vw   = (const float*)d_in[2];
  const float* Vb   = (const float*)d_in[3];
  const float* Ww   = (const float*)d_in[4];
  const float* Wb   = (const float*)d_in[5];
  float*       out  = (float*)d_out;
  _Float16*    Vh   = (_Float16*)d_ws;   // 2 MB f16 copy of V_w

  // D*D / (256 threads * 4 elems) = 1024 blocks
  convert_vw<<<(D * D) / (256 * 4), 256, 0, stream>>>(Vw, Vh);
  rnn_scan<<<BATCH / BT, 512, 0, stream>>>(x, Uemb, Vb, Vh, Ww, Wb, out);
}
